// CRELayer_37374805410127
// MI455X (gfx1250) — matrix operand order repaired and checked by reading, not compiled, not run on hardware
//
#include <hip/hip_runtime.h>
#include <hip/hip_bf16.h>
#include <math.h>

typedef __bf16 bf16;
typedef __attribute__((ext_vector_type(16))) __bf16 v16bf;
typedef __attribute__((ext_vector_type(8)))  __bf16 v8bf;
typedef __attribute__((ext_vector_type(8)))  float  v8f;
typedef int v4i __attribute__((vector_size(16)));     // builtin's b128 unit

#define NB   4
#define TT   8192
#define DD   1024
#define BT   (NB * TT)        // 32768 rows
#define KP   40               // padded LDS row (elements) -> 80B stride, conflict-free
#define NCH  32               // scan chunks
#define CHK  (TT / NCH)       // 256

// Async LDS-direct copy path (gfx1250): guarded so compile never fails.
#if defined(__has_builtin)
#if __has_builtin(__builtin_amdgcn_global_load_async_to_lds_b128) && \
    __has_builtin(__builtin_amdgcn_s_wait_asynccnt)
#define HAVE_ASYNC_LDS 1
#endif
#endif
#ifndef HAVE_ASYNC_LDS
#define HAVE_ASYNC_LDS 0
#endif

#if HAVE_ASYNC_LDS
#define ASYNC_CP16(g, l, off)                                            \
  __builtin_amdgcn_global_load_async_to_lds_b128(                        \
      (__attribute__((address_space(1))) v4i*)(g),                       \
      (__attribute__((address_space(3))) v4i*)(l), (off), 0)
#define ASYNC_WAIT() __builtin_amdgcn_s_wait_asynccnt(0)
#endif

// ---------------------------------------------------------------------------
// LayerNorm: f32 x[BT, D] -> bf16 normed[BT, D]
// ---------------------------------------------------------------------------
__global__ __launch_bounds__(256) void ln_kernel(const float* __restrict__ x,
                                                 const float* __restrict__ gamma,
                                                 const float* __restrict__ beta,
                                                 bf16* __restrict__ out) {
  const int row = blockIdx.x;
  const int tid = threadIdx.x;
  __shared__ float red0[8];
  __shared__ float red1[8];

  const float4 v = ((const float4*)(x + (size_t)row * DD))[tid];

  float s = v.x + v.y + v.z + v.w;
  for (int off = 16; off > 0; off >>= 1) s += __shfl_down(s, off, 32);
  if ((tid & 31) == 0) red0[tid >> 5] = s;
  __syncthreads();
  float tot = 0.f;
  #pragma unroll
  for (int i = 0; i < 8; i++) tot += red0[i];
  const float mean = tot * (1.0f / DD);

  float dx0 = v.x - mean, dx1 = v.y - mean, dx2 = v.z - mean, dx3 = v.w - mean;
  float q = dx0 * dx0 + dx1 * dx1 + dx2 * dx2 + dx3 * dx3;
  for (int off = 16; off > 0; off >>= 1) q += __shfl_down(q, off, 32);
  if ((tid & 31) == 0) red1[tid >> 5] = q;
  __syncthreads();
  float qtot = 0.f;
  #pragma unroll
  for (int i = 0; i < 8; i++) qtot += red1[i];
  const float rstd = rsqrtf(qtot * (1.0f / DD) + 1e-5f);

  const float4 ga = ((const float4*)gamma)[tid];
  const float4 be = ((const float4*)beta)[tid];
  bf16* o = out + (size_t)row * DD + tid * 4;
  o[0] = (bf16)(dx0 * rstd * ga.x + be.x);
  o[1] = (bf16)(dx1 * rstd * ga.y + be.y);
  o[2] = (bf16)(dx2 * rstd * ga.z + be.z);
  o[3] = (bf16)(dx3 * rstd * ga.w + be.w);
}

// ---------------------------------------------------------------------------
// f32 -> bf16 weight convert
// ---------------------------------------------------------------------------
__global__ __launch_bounds__(256) void cvt_kernel(const float* __restrict__ a,
                                                  bf16* __restrict__ b, int n) {
  int i = blockIdx.x * 256 + threadIdx.x;
  if (i < n) b[i] = (bf16)a[i];
}

// ---------------------------------------------------------------------------
// Shared GEMM helpers: A fragment gather, contiguous k = 16*half + s (16-bit A 16x32)
// ---------------------------------------------------------------------------
__device__ __forceinline__ void load_afrags(const bf16* tile, int wm, int l16,
                                            int half, v16bf af[2]) {
  #pragma unroll
  for (int mi = 0; mi < 2; mi++) {
    const bf16* base = tile + (wm * 32 + mi * 16 + l16) * KP;
    union { v16bf v; v8bf h[2]; } u;
    u.h[0] = *(const v8bf*)(base + half * 16);
    u.h[1] = *(const v8bf*)(base + half * 16 + 8);
    af[mi] = u.v;
  }
}

// ---------------------------------------------------------------------------
// Dual GEMM + gate fusion:  s[t,e] = (2*sigmoid(N·Wg^T + bg) - 1) * (N·Wv^T + bv)
// Block: 256 threads = 8 waves (4 in M, 2 in N). Tile BM=128, BN=64.
// A tile double-buffered in LDS via async LDS-direct copies when available.
// ---------------------------------------------------------------------------
__global__ __launch_bounds__(256) void gemm_gv_kernel(
    const bf16* __restrict__ A,                 // [BT, D] normed
    const bf16* __restrict__ Wg,                // [D, D] bf16 (row e, col d)
    const bf16* __restrict__ Wv,                // [D, D] bf16
    const float* __restrict__ bg, const float* __restrict__ bv,
    float* __restrict__ S)                      // [BT, D] f32
{
#if HAVE_ASYNC_LDS
  __shared__ bf16 lds_a[2][128 * KP];
#else
  __shared__ bf16 lds_a[1][128 * KP];
#endif

  const int tid  = threadIdx.x;
  const int wave = tid >> 5;
  const int lane = tid & 31;
  const int wm   = wave & 3;          // wave position in M
  const int wn   = wave >> 2;         // wave position in N
  const int m0   = blockIdx.x * 128;
  const int n0   = blockIdx.y * 64 + wn * 32;
  const int half = lane >> 4;
  const int l16  = lane & 15;

  const v8f zf = {0.f, 0.f, 0.f, 0.f, 0.f, 0.f, 0.f, 0.f};
  v8f cg[2][2] = {{zf, zf}, {zf, zf}};
  v8f cv[2][2] = {{zf, zf}, {zf, zf}};

  // A-tile staging: each thread moves 32B (two b128) per K step
  const int   arow = tid >> 1;
  const int   acol = (tid & 1) * 16;
  const bf16* aG   = A + (size_t)(m0 + arow) * DD + acol;

#if HAVE_ASYNC_LDS
  bf16* aL0 = &lds_a[0][arow * KP + acol];
  bf16* aL1 = &lds_a[1][arow * KP + acol];

  ASYNC_CP16(aG, aL0, 0);             // preload K-tile 0 into buffer 0
  ASYNC_CP16(aG, aL0, 32);
  ASYNC_WAIT();
  __syncthreads();

  int buf = 0;
  for (int k = 0; k < DD; k += 32) {
    if (k + 32 < DD) {                // fire next tile into the other buffer
      const bf16* g = aG + k + 32;
      bf16*       l = buf ? aL0 : aL1;
      ASYNC_CP16(g, l, 0);
      ASYNC_CP16(g, l, 32);
    }
    const bf16* tile = buf ? lds_a[1] : lds_a[0];
#else
  bf16* aL = &lds_a[0][arow * KP + acol];
  for (int k = 0; k < DD; k += 32) {
    const uint4* src = (const uint4*)(aG + k);
    uint4 t0 = src[0];
    uint4 t1 = src[1];
    ((uint4*)aL)[0] = t0;
    ((uint4*)aL)[1] = t1;
    __syncthreads();
    if (k + 32 < DD) __builtin_prefetch(aG + k + 32, 0, 3);
    const bf16* tile = lds_a[0];
#endif

    v16bf afrag[2];
    load_afrags(tile, wm, l16, half, afrag);

    // B fragments straight from global (weights stay L2-resident):
    // lane<16 -> n=lane, k..k+15 ; lane>=16 -> n=lane-16, k+16..k+31
    v16bf bfg[2], bfv[2];
    #pragma unroll
    for (int ni = 0; ni < 2; ni++) {
      const size_t off = (size_t)(n0 + ni * 16 + l16) * DD + k + half * 16;
      bfg[ni] = *(const v16bf*)(Wg + off);
      bfv[ni] = *(const v16bf*)(Wv + off);
    }
    #pragma unroll
    for (int mi = 0; mi < 2; mi++)
      #pragma unroll
      for (int ni = 0; ni < 2; ni++) {
        cg[mi][ni] = __builtin_amdgcn_wmma_f32_16x16x32_bf16(
            false, afrag[mi], false, bfg[ni], (short)0, cg[mi][ni], false, false);
        cv[mi][ni] = __builtin_amdgcn_wmma_f32_16x16x32_bf16(
            false, afrag[mi], false, bfv[ni], (short)0, cv[mi][ni], false, false);
      }

#if HAVE_ASYNC_LDS
    ASYNC_WAIT();
    __syncthreads();
    buf ^= 1;
#else
    __syncthreads();
#endif
  }

  // Epilogue: C/D layout -> lane: n = l16, vgpr r: m = half*8 + r
  #pragma unroll
  for (int mi = 0; mi < 2; mi++)
    #pragma unroll
    for (int ni = 0; ni < 2; ni++) {
      const int e = n0 + ni * 16 + l16;
      const float bge = bg[e], bve = bv[e];
      #pragma unroll
      for (int r = 0; r < 8; r++) {
        const int t = m0 + wm * 32 + mi * 16 + half * 8 + r;
        const float gval = cg[mi][ni][r] + bge;
        const float vval = cv[mi][ni][r] + bve;
        const float sig  = 1.0f / (1.0f + __expf(-gval));
        S[(size_t)t * DD + e] = (2.0f * sig - 1.0f) * vval;
      }
    }
}

// ---------------------------------------------------------------------------
// Scan pass 1: per-chunk sums of s over T.  partial[b, c, d] = sum of chunk c
// ---------------------------------------------------------------------------
__global__ __launch_bounds__(256) void scan_chunk_kernel(const float* __restrict__ S,
                                                         float* __restrict__ partial) {
  const int d = blockIdx.x * 256 + threadIdx.x;
  const int c = blockIdx.y;
  const int b = blockIdx.z;
  const size_t base = ((size_t)b * TT + (size_t)c * CHK) * DD + d;
  float acc = 0.f;
  for (int i = 0; i < CHK; i++) acc += S[base + (size_t)i * DD];
  partial[((size_t)b * NCH + c) * DD + d] = acc;
}

// Scan pass 2: exclusive prefix over the 32 chunk sums (in place)
__global__ __launch_bounds__(256) void scan_offsets_kernel(float* __restrict__ partial) {
  const int d = blockIdx.x * 256 + threadIdx.x;
  const int b = blockIdx.y;
  float run = 0.f;
  for (int c = 0; c < NCH; c++) {
    const size_t idx = ((size_t)b * NCH + c) * DD + d;
    const float v = partial[idx];
    partial[idx] = run;
    run += v;
  }
}

// Scan pass 3: running cumsum within chunk + exp decay, emit bf16 u
__global__ __launch_bounds__(256) void scan_apply_kernel(const float* __restrict__ S,
                                                         const float* __restrict__ partial,
                                                         const float* __restrict__ log_decay,
                                                         bf16* __restrict__ U) {
  const int d = blockIdx.x * 256 + threadIdx.x;
  const int c = blockIdx.y;
  const int b = blockIdx.z;
  float acc = partial[((size_t)b * NCH + c) * DD + d];
  const float alpha = log1pf(__expf(log_decay[0]));   // softplus
  const int t0 = c * CHK;
  const size_t base = ((size_t)b * TT + t0) * DD + d;
  for (int i = 0; i < CHK; i++) {
    acc += S[base + (size_t)i * DD];
    const float dec = __expf(-alpha * (float)(t0 + i));
    U[base + (size_t)i * DD] = (bf16)(acc * dec);
  }
}

// ---------------------------------------------------------------------------
// Output GEMM: out = x + U·Wo^T + bo   (same tiling as gemm_gv, one weight)
// ---------------------------------------------------------------------------
__global__ __launch_bounds__(256) void gemm_o_kernel(
    const bf16* __restrict__ U, const bf16* __restrict__ Wo,
    const float* __restrict__ bo, const float* __restrict__ x,
    float* __restrict__ out)
{
#if HAVE_ASYNC_LDS
  __shared__ bf16 lds_a[2][128 * KP];
#else
  __shared__ bf16 lds_a[1][128 * KP];
#endif

  const int tid  = threadIdx.x;
  const int wave = tid >> 5;
  const int lane = tid & 31;
  const int wm   = wave & 3;
  const int wn   = wave >> 2;
  const int m0   = blockIdx.x * 128;
  const int n0   = blockIdx.y * 64 + wn * 32;
  const int half = lane >> 4;
  const int l16  = lane & 15;

  const v8f zf = {0.f, 0.f, 0.f, 0.f, 0.f, 0.f, 0.f, 0.f};
  v8f co[2][2] = {{zf, zf}, {zf, zf}};

  const int   arow = tid >> 1;
  const int   acol = (tid & 1) * 16;
  const bf16* aG   = U + (size_t)(m0 + arow) * DD + acol;

#if HAVE_ASYNC_LDS
  bf16* aL0 = &lds_a[0][arow * KP + acol];
  bf16* aL1 = &lds_a[1][arow * KP + acol];

  ASYNC_CP16(aG, aL0, 0);
  ASYNC_CP16(aG, aL0, 32);
  ASYNC_WAIT();
  __syncthreads();

  int buf = 0;
  for (int k = 0; k < DD; k += 32) {
    if (k + 32 < DD) {
      const bf16* g = aG + k + 32;
      bf16*       l = buf ? aL0 : aL1;
      ASYNC_CP16(g, l, 0);
      ASYNC_CP16(g, l, 32);
    }
    const bf16* tile = buf ? lds_a[1] : lds_a[0];
#else
  bf16* aL = &lds_a[0][arow * KP + acol];
  for (int k = 0; k < DD; k += 32) {
    const uint4* src = (const uint4*)(aG + k);
    uint4 t0 = src[0];
    uint4 t1 = src[1];
    ((uint4*)aL)[0] = t0;
    ((uint4*)aL)[1] = t1;
    __syncthreads();
    if (k + 32 < DD) __builtin_prefetch(aG + k + 32, 0, 3);
    const bf16* tile = lds_a[0];
#endif

    v16bf afrag[2];
    load_afrags(tile, wm, l16, half, afrag);

    v16bf bfo[2];
    #pragma unroll
    for (int ni = 0; ni < 2; ni++) {
      const size_t off = (size_t)(n0 + ni * 16 + l16) * DD + k + half * 16;
      bfo[ni] = *(const v16bf*)(Wo + off);
    }
    #pragma unroll
    for (int mi = 0; mi < 2; mi++)
      #pragma unroll
      for (int ni = 0; ni < 2; ni++)
        co[mi][ni] = __builtin_amdgcn_wmma_f32_16x16x32_bf16(
            false, afrag[mi], false, bfo[ni], (short)0, co[mi][ni], false, false);

#if HAVE_ASYNC_LDS
    ASYNC_WAIT();
    __syncthreads();
    buf ^= 1;
#else
    __syncthreads();
#endif
  }

  #pragma unroll
  for (int mi = 0; mi < 2; mi++)
    #pragma unroll
    for (int ni = 0; ni < 2; ni++) {
      const int e = n0 + ni * 16 + l16;
      const float boe = bo[e];
      #pragma unroll
      for (int r = 0; r < 8; r++) {
        const int t = m0 + wm * 32 + mi * 16 + half * 8 + r;
        const size_t idx = (size_t)t * DD + e;
        out[idx] = x[idx] + co[mi][ni][r] + boe;
      }
    }
}

// ---------------------------------------------------------------------------
extern "C" void kernel_launch(void* const* d_in, const int* in_sizes, int n_in,
                              void* d_out, int out_size, void* d_ws, size_t ws_size,
                              hipStream_t stream) {
  const float* x        = (const float*)d_in[0];
  const float* ln_gamma = (const float*)d_in[1];
  const float* ln_beta  = (const float*)d_in[2];
  const float* Wg       = (const float*)d_in[3];
  const float* bg       = (const float*)d_in[4];
  const float* Wv       = (const float*)d_in[5];
  const float* bv       = (const float*)d_in[6];
  const float* Wo       = (const float*)d_in[7];
  const float* bo       = (const float*)d_in[8];
  const float* log_dec  = (const float*)d_in[9];
  float* out            = (float*)d_out;

  // Workspace layout
  char* ws = (char*)d_ws;
  const size_t SZ_ACT_BF16 = (size_t)BT * DD * sizeof(bf16);   // 64 MB
  const size_t SZ_S_F32    = (size_t)BT * DD * sizeof(float);  // 128 MB
  const size_t SZ_W_BF16   = (size_t)DD * DD * sizeof(bf16);   // 2 MB

  bf16*  normed  = (bf16*)(ws);                       // reused as U after gemm_gv
  float* Sbuf    = (float*)(ws + SZ_ACT_BF16);
  bf16*  WgB     = (bf16*)(ws + SZ_ACT_BF16 + SZ_S_F32);
  bf16*  WvB     = (bf16*)(ws + SZ_ACT_BF16 + SZ_S_F32 + SZ_W_BF16);
  bf16*  WoB     = (bf16*)(ws + SZ_ACT_BF16 + SZ_S_F32 + 2 * SZ_W_BF16);
  float* partial = (float*)(ws + SZ_ACT_BF16 + SZ_S_F32 + 3 * SZ_W_BF16);
  bf16*  U       = normed;                            // alias: normed dead after gemm_gv

  // 1) LayerNorm -> bf16
  ln_kernel<<<BT, 256, 0, stream>>>(x, ln_gamma, ln_beta, normed);

  // 2) Weights -> bf16 (L2-resident thereafter)
  const int wn = DD * DD;
  cvt_kernel<<<(wn + 255) / 256, 256, 0, stream>>>(Wg, WgB, wn);
  cvt_kernel<<<(wn + 255) / 256, 256, 0, stream>>>(Wv, WvB, wn);
  cvt_kernel<<<(wn + 255) / 256, 256, 0, stream>>>(Wo, WoB, wn);

  // 3) Fused dual GEMM + gate -> s (f32)
  gemm_gv_kernel<<<dim3(BT / 128, DD / 64), 256, 0, stream>>>(
      normed, WgB, WvB, bg, bv, Sbuf);

  // 4) Chunked scan over T, fused with decay -> u (bf16)
  scan_chunk_kernel  <<<dim3(DD / 256, NCH, NB), 256, 0, stream>>>(Sbuf, partial);
  scan_offsets_kernel<<<dim3(DD / 256, NB),      256, 0, stream>>>(partial);
  scan_apply_kernel  <<<dim3(DD / 256, NCH, NB), 256, 0, stream>>>(Sbuf, partial, log_dec, U);

  // 5) Output GEMM + bias + residual
  gemm_o_kernel<<<dim3(BT / 128, DD / 64), 256, 0, stream>>>(U, WoB, bo, x, out);
}
